// graph_MPNN_44719199486483
// MI455X (gfx1250) — matrix operand order repaired and checked by reading, not compiled, not run on hardware
//
#include <hip/hip_runtime.h>
#include <hip/hip_bf16.h>
#include <cfloat>

// ---------------- sizes (fixed by reference) ----------------
#define NN      16384
#define EE      262144
#define ET      (EE + NN)          // 278528 edges incl self loops
#define FIN_N   768
#define FIN_E   64
#define FF      256
#define FE      128
#define F2      512
#define HH      8
#define CC      32
#define LL      2

typedef __attribute__((ext_vector_type(16))) __bf16 v16bf;
typedef __attribute__((ext_vector_type(8)))  float  v8f;
typedef unsigned int u32x4 __attribute__((ext_vector_type(4)));
typedef unsigned int u32x8 __attribute__((ext_vector_type(8)));

union BFrag { v16bf v; unsigned short s[16]; unsigned u[8]; };
union F8 { float4 v[2]; float f[8]; };

__device__ __forceinline__ unsigned short f2bf(float f) {
  unsigned u = __float_as_uint(f);
  unsigned r = u + 0x7fffu + ((u >> 16) & 1u);   // round-to-nearest-even
  return (unsigned short)(r >> 16);
}

// monotonic float <-> uint encoding for atomic max over signed floats
__device__ __forceinline__ unsigned encf(float f) {
  unsigned u = __float_as_uint(f);
  return (u & 0x80000000u) ? ~u : (u | 0x80000000u);
}
__device__ __forceinline__ float decf(unsigned u) {
  unsigned v = (u & 0x80000000u) ? (u & 0x7fffffffu) : ~u;
  return __uint_as_float(v);
}

// ======================================================================
// WMMA bf16 GEMM: C[M,N] = A[M,K](fp32, cvt on load) * W[N,K](bf16)
// One wave computes a 16x64 strip (4 accumulators) -> A fragment and its
// bf16 conversion amortized over 4 WMMAs; K % 64 == 0, N % 64 == 0.
// epi: 0 = none, 1 = +bias[n], 2 = exact GELU
// ======================================================================
__global__ __launch_bounds__(128) void gat_wmma_gemm(
    const float* __restrict__ A, const unsigned short* __restrict__ B,
    const float* __restrict__ bias, float* __restrict__ C,
    int M, int N, int K, int epi)
{
  const int lane = threadIdx.x & 31;
  const int wave = threadIdx.x >> 5;
  const int nStrips = N >> 6;                   // 64-wide strips
  const int tile = blockIdx.x * 4 + wave;
  const int nTiles = (M >> 4) * nStrips;
  if (tile >= nTiles) return;

  const int mt = tile / nStrips;
  const int ns = tile - mt * nStrips;
  const int m0 = mt << 4;
  const int n0 = ns << 6;
  const int half = lane >> 4;       // 0: lanes 0-15, 1: lanes 16-31
  const int l16  = lane & 15;

  const float4* __restrict__ Arow = (const float4*)(A + (long)(m0 + l16) * K);
  const uint4* Brow[4];
#pragma unroll
  for (int nn = 0; nn < 4; ++nn)
    Brow[nn] = (const uint4*)(B + (long)(n0 + nn * 16 + l16) * K);

  v8f acc[4] = {{}, {}, {}, {}};
#pragma unroll 2
  for (int k0 = 0; k0 < K; k0 += 32) {
    // A 16x32 bf16 fragment: lanes 0-15 hold K{0..7,16..23}, lanes 16-31 K{8..15,24..31}
    const int ka = (k0 + 8 * half) >> 2;        // float4 index; offsets ka, ka+16 floats
    F8 lo, hi;
    lo.v[0] = Arow[ka];     lo.v[1] = Arow[ka + 1];
    hi.v[0] = Arow[ka + 4]; hi.v[1] = Arow[ka + 5];
    BFrag a;
#pragma unroll
    for (int j = 0; j < 8; ++j) { a.s[j] = f2bf(lo.f[j]); a.s[8 + j] = f2bf(hi.f[j]); }

    const int kb = (k0 + 8 * half) >> 3;        // uint4 index (8 ushorts each)
#pragma unroll
    for (int nn = 0; nn < 4; ++nn) {
      BFrag b;
      uint4 b0 = Brow[nn][kb], b1 = Brow[nn][kb + 2];
      b.u[0] = b0.x; b.u[1] = b0.y; b.u[2] = b0.z; b.u[3] = b0.w;
      b.u[4] = b1.x; b.u[5] = b1.y; b.u[6] = b1.z; b.u[7] = b1.w;
      acc[nn] = __builtin_amdgcn_wmma_f32_16x16x32_bf16(
          false, a.v, false, b.v, (short)0, acc[nn], false, false);
    }
  }

  const int mbase = m0 + 8 * half;   // D layout: VGPR i -> row i (+8 for upper lanes)
#pragma unroll
  for (int nn = 0; nn < 4; ++nn) {
    const int col = n0 + nn * 16 + l16;
    const float bv = (epi == 1) ? bias[col] : 0.0f;
#pragma unroll
    for (int i = 0; i < 8; ++i) {
      float v = acc[nn][i] + bv;
      if (epi == 2) v = 0.5f * v * (1.0f + erff(v * 0.70710678118654752f));
      C[(long)(mbase + i) * N + col] = v;
    }
  }
}

// ---------------- weight convert: element-wise fp32 -> bf16 ([N,K] kept) --
__global__ void gat_cvt_bf16(const float* __restrict__ W, unsigned short* __restrict__ Bt,
                             long n)
{
  long i = (long)blockIdx.x * blockDim.x + threadIdx.x;
  if (i < n) Bt[i] = f2bf(W[i]);
}

// ---------------- utility fills ------------------------------------------
__global__ void gat_zero_f32(float* p, long n) {
  long i = (long)blockIdx.x * blockDim.x + threadIdx.x;
  if (i < n) p[i] = 0.0f;
}
__global__ void gat_fill_u32(unsigned* p, unsigned v, long n) {
  long i = (long)blockIdx.x * blockDim.x + threadIdx.x;
  if (i < n) p[i] = v;
}

// ---------------- self-loop mean edge attr --------------------------------
__global__ __launch_bounds__(128) void gat_loopsum(
    const int* __restrict__ eidx, const float* __restrict__ ea,
    float* __restrict__ loop, float* __restrict__ cnt)
{
  const int e = blockIdx.x;          // one block (128 thr) per edge
  const int c = threadIdx.x;
  const int dst = eidx[EE + e];
  atomicAdd(&loop[(long)dst * FE + c], ea[(long)e * FE + c]);
  if (c == 0) atomicAdd(&cnt[dst], 1.0f);
}
__global__ void gat_loopdiv(float* __restrict__ loop, const float* __restrict__ cnt) {
  int i = blockIdx.x * blockDim.x + threadIdx.x;
  if (i >= NN * FE) return;
  loop[i] /= fmaxf(cnt[i >> 7], 1.0f);
}

// ---------------- per-edge logits + segment max (wave per edge) -----------
__global__ __launch_bounds__(256) void gat_logits(
    const float* __restrict__ xl, const float* __restrict__ xr,
    const float* __restrict__ ee, const int* __restrict__ eidx,
    const float* __restrict__ att, float* __restrict__ logits,
    unsigned* __restrict__ maxb)
{
  const int wave = threadIdx.x >> 5, lane = threadIdx.x & 31;
  const long e = (long)blockIdx.x * 8 + wave;
  int src, dst;
  if (e < EE) { src = eidx[e]; dst = eidx[EE + e]; }
  else        { src = dst = (int)(e - EE); }

  const int c0 = lane * 8;           // 8 channels per lane, all in head c0/32
  const int h  = lane >> 2;
  const float4* pl = (const float4*)(xl + (long)src * FF + c0);
  const float4* pr = (const float4*)(xr + (long)dst * FF + c0);
  const float4* pe = (const float4*)(ee + e * FF + c0);
  const float4* pa = (const float4*)(att + c0);

  F8 vl, vr, ve, va;
  vl.v[0] = pl[0]; vl.v[1] = pl[1];
  vr.v[0] = pr[0]; vr.v[1] = pr[1];
  ve.v[0] = pe[0]; ve.v[1] = pe[1];
  va.v[0] = pa[0]; va.v[1] = pa[1];

  float partial = 0.0f;
#pragma unroll
  for (int j = 0; j < 8; ++j) {
    float m = vl.f[j] + vr.f[j] + ve.f[j];
    float g = (m >= 0.0f) ? m : 0.2f * m;      // leaky_relu(0.2)
    partial += g * va.f[j];
  }
  partial += __shfl_xor(partial, 1, 32);
  partial += __shfl_xor(partial, 2, 32);
  if ((lane & 3) == 0) {
    logits[e * HH + h] = partial;
    atomicMax(&maxb[(long)dst * HH + h], encf(partial));
  }
}

// ---------------- exp(logit - max) + segment sum --------------------------
__global__ void gat_expsum(const int* __restrict__ eidx, float* __restrict__ num,
                           const unsigned* __restrict__ maxb, float* __restrict__ sumb)
{
  long i = (long)blockIdx.x * blockDim.x + threadIdx.x;
  if (i >= (long)ET * HH) return;
  long e = i >> 3; int h = (int)(i & 7);
  int dst = (e < EE) ? eidx[EE + e] : (int)(e - EE);
  float m = decf(maxb[(long)dst * HH + h]);
  float v = expf(num[i] - m);
  num[i] = v;
  atomicAdd(&sumb[(long)dst * HH + h], v);
}

// ---------------- weighted scatter aggregation ----------------------------
__global__ __launch_bounds__(256) void gat_aggregate(
    const int* __restrict__ eidx, const float* __restrict__ num,
    const float* __restrict__ sumb, const float* __restrict__ xl,
    float* __restrict__ agg)
{
  const long e = blockIdx.x;         // one block (256 thr = F channels) per edge
  const int c = threadIdx.x;
  const int h = c >> 5;
  int src, dst;
  if (e < EE) { src = eidx[e]; dst = eidx[EE + e]; }
  else        { src = dst = (int)(e - EE); }
  const float* psrc = xl + (long)src * FF;
  __builtin_prefetch(psrc + c, 0, 1);          // global_prefetch_b8
  float alpha = num[e * HH + h] / (sumb[(long)dst * HH + h] + 1e-16f);
  atomicAdd(&agg[(long)dst * FF + c], alpha * psrc[c]);
}

// ---------------- fused residual + LayerNorm over F=256 -------------------
__global__ __launch_bounds__(256) void gat_ln(
    const float* __restrict__ base, const float* __restrict__ r,
    const float* __restrict__ cbias, const float* __restrict__ g,
    const float* __restrict__ b, float* __restrict__ out)
{
  __shared__ float red[256];
  const int n = blockIdx.x, c = threadIdx.x;
  float v = base[(long)n * FF + c] + r[(long)n * FF + c] + (cbias ? cbias[c] : 0.0f);
  red[c] = v; __syncthreads();
  for (int s = 128; s > 0; s >>= 1) { if (c < s) red[c] += red[c + s]; __syncthreads(); }
  float mu = red[0] * (1.0f / FF); __syncthreads();
  float d = v - mu; red[c] = d * d; __syncthreads();
  for (int s = 128; s > 0; s >>= 1) { if (c < s) red[c] += red[c + s]; __syncthreads(); }
  float var = red[0] * (1.0f / FF);
  out[(long)n * FF + c] = d * rsqrtf(var + 1e-5f) * g[c] + b[c];
}

// ======================================================================
// Final copy via Tensor Data Mover: global -> LDS (TDM) -> global.
// D# built per CDNA5 ISA (1-D tile, 4B elts).
// ======================================================================
#define TDM_CHUNK 4096   // floats per block (16 KB)

__global__ __launch_bounds__(256) void gat_tdm_copy(const float* __restrict__ src,
                                                    float* __restrict__ dst)
{
  __shared__ float buf[TDM_CHUNK];
  const long base = (long)blockIdx.x * TDM_CHUNK;

  if (threadIdx.x < 32) {            // wave 0 issues the tensor DMA
    unsigned long long ga = (unsigned long long)(const void*)(src + base);
    u32x4 g0;
    g0[0] = (unsigned)__builtin_amdgcn_readfirstlane(1);               // count=1
    g0[1] = (unsigned)__builtin_amdgcn_readfirstlane(0);               // lds_addr=0
    g0[2] = (unsigned)__builtin_amdgcn_readfirstlane((int)(ga & 0xffffffffu));
    g0[3] = (unsigned)__builtin_amdgcn_readfirstlane(
                (int)(((ga >> 32) & 0x01ffffffu) | 0x80000000u));      // addr[56:32] | type=2
    u32x8 g1;
    g1[0] = (unsigned)__builtin_amdgcn_readfirstlane(0x00020000);      // data_size=4B
    g1[1] = (unsigned)__builtin_amdgcn_readfirstlane(TDM_CHUNK << 16); // tensor_dim0 lo16
    g1[2] = (unsigned)__builtin_amdgcn_readfirstlane(1 << 16);         // tensor_dim1=1
    g1[3] = (unsigned)__builtin_amdgcn_readfirstlane(TDM_CHUNK << 16); // tile_dim0
    g1[4] = (unsigned)__builtin_amdgcn_readfirstlane(0);               // tile_dim1/2 = 0
    g1[5] = (unsigned)__builtin_amdgcn_readfirstlane(TDM_CHUNK);       // dim0_stride lo32
    g1[6] = (unsigned)__builtin_amdgcn_readfirstlane(0);
    g1[7] = (unsigned)__builtin_amdgcn_readfirstlane(0);
    asm volatile("tensor_load_to_lds %0, %1" : : "s"(g0), "s"(g1) : "memory");
    __builtin_amdgcn_s_wait_tensorcnt(0);
  }
  __syncthreads();
#pragma unroll
  for (int j = 0; j < TDM_CHUNK / 256; ++j) {
    int i = threadIdx.x + j * 256;
    dst[base + i] = buf[i];
  }
}

// ======================================================================
// host driver
// ======================================================================
static inline void launch_gemm(const float* A, const unsigned short* B, const float* bias,
                               float* C, int M, int N, int K, int epi, hipStream_t s)
{
  int tiles = (M >> 4) * (N >> 6);
  gat_wmma_gemm<<<(tiles + 3) / 4, 128, 0, s>>>(A, B, bias, C, M, N, K, epi);
}

extern "C" void kernel_launch(void* const* d_in, const int* in_sizes, int n_in,
                              void* d_out, int out_size, void* d_ws, size_t ws_size,
                              hipStream_t stream)
{
  (void)in_sizes; (void)n_in; (void)out_size; (void)ws_size;
  const float* node_feats = (const float*)d_in[0];
  const int*   eidx       = (const int*)  d_in[1];
  const float* edge_attr  = (const float*)d_in[2];
  const float* init_node_w = (const float*)d_in[3];
  const float* init_edge_w = (const float*)d_in[4];
  const float* lin_l_w = (const float*)d_in[5];
  const float* lin_l_b = (const float*)d_in[6];
  const float* lin_r_w = (const float*)d_in[7];
  const float* lin_r_b = (const float*)d_in[8];
  const float* lin_edge_w = (const float*)d_in[9];
  const float* att      = (const float*)d_in[10];
  const float* gat_bias = (const float*)d_in[11];
  const float* ln1_g = (const float*)d_in[12];
  const float* ln1_b = (const float*)d_in[13];
  const float* mlp_w1 = (const float*)d_in[14];
  const float* mlp_w2 = (const float*)d_in[15];
  const float* ln2_g = (const float*)d_in[16];
  const float* ln2_b = (const float*)d_in[17];

  // ---- workspace carve-out (256B aligned) ----
  char* wp = (char*)d_ws;
  auto alloc = [&](size_t bytes) -> void* {
    void* p = (void*)wp;
    wp += (bytes + 255) & ~(size_t)255;
    return p;
  };
  float* x      = (float*)alloc((size_t)NN * FF * 4);
  float* ea_all = (float*)alloc((size_t)ET * FE * 4);   // [E edges | N loop rows]
  float* e_emb  = (float*)alloc((size_t)ET * FF * 4);
  float* xl     = (float*)alloc((size_t)NN * FF * 4);
  float* xr     = (float*)alloc((size_t)NN * FF * 4);
  float* num    = (float*)alloc((size_t)ET * HH * 4);   // logits -> exp numerators
  unsigned* maxb = (unsigned*)alloc((size_t)NN * HH * 4);
  float* sumb   = (float*)alloc((size_t)NN * HH * 4);
  float* agg    = (float*)alloc((size_t)NN * FF * 4);
  float* hbuf   = (float*)alloc((size_t)NN * FF * 4);
  float* mid    = (float*)alloc((size_t)NN * F2 * 4);
  float* ybuf   = (float*)alloc((size_t)NN * FF * 4);
  float* cnt    = (float*)alloc((size_t)NN * 4);
  unsigned short* Bn  = (unsigned short*)alloc((size_t)FF * FIN_N * 2);
  unsigned short* Be  = (unsigned short*)alloc((size_t)FE * FIN_E * 2);
  unsigned short* Bll = (unsigned short*)alloc((size_t)LL * FF * FF * 2);
  unsigned short* Blr = (unsigned short*)alloc((size_t)LL * FF * FF * 2);
  unsigned short* Ble = (unsigned short*)alloc((size_t)LL * FF * FE * 2);
  unsigned short* B1  = (unsigned short*)alloc((size_t)LL * F2 * FF * 2);
  unsigned short* B2  = (unsigned short*)alloc((size_t)LL * FF * F2 * 2);

  auto cvt = [&](const float* W, unsigned short* Bt, long tot) {
    gat_cvt_bf16<<<(unsigned)((tot + 255) / 256), 256, 0, stream>>>(W, Bt, tot);
  };

  // ---- weight conversion (fp32 -> bf16, layout [N,K] preserved) ----
  cvt(init_node_w, Bn, (long)FF * FIN_N);
  cvt(init_edge_w, Be, (long)FE * FIN_E);
  cvt(lin_l_w,    Bll, (long)LL * FF * FF);
  cvt(lin_r_w,    Blr, (long)LL * FF * FF);
  cvt(lin_edge_w, Ble, (long)LL * FF * FE);
  cvt(mlp_w1,     B1,  (long)LL * F2 * FF);
  cvt(mlp_w2,     B2,  (long)LL * FF * F2);

  // ---- input projections ----
  launch_gemm(node_feats, Bn, nullptr, x, NN, FF, FIN_N, 0, stream);
  launch_gemm(edge_attr, Be, nullptr, ea_all, EE, FE, FIN_E, 0, stream);

  // ---- self-loop edge attr = mean of incoming ----
  {
    long nz = (long)NN * FE;
    gat_zero_f32<<<(unsigned)((nz + 255) / 256), 256, 0, stream>>>(ea_all + (size_t)EE * FE, nz);
    gat_zero_f32<<<(NN + 255) / 256, 256, 0, stream>>>(cnt, NN);
    gat_loopsum<<<EE, FE, 0, stream>>>(eidx, ea_all, ea_all + (size_t)EE * FE, cnt);
    gat_loopdiv<<<(NN * FE + 255) / 256, 256, 0, stream>>>(ea_all + (size_t)EE * FE, cnt);
  }

  // ---- GATv2 layers ----
  for (int l = 0; l < LL; ++l) {
    launch_gemm(ea_all, Ble + (size_t)l * FF * FE, nullptr, e_emb, ET, FF, FE, 0, stream);
    launch_gemm(x, Bll + (size_t)l * FF * FF, lin_l_b + (size_t)l * FF, xl, NN, FF, FF, 1, stream);
    launch_gemm(x, Blr + (size_t)l * FF * FF, lin_r_b + (size_t)l * FF, xr, NN, FF, FF, 1, stream);

    long nh = (long)NN * HH;
    gat_fill_u32<<<(unsigned)((nh + 255) / 256), 256, 0, stream>>>(maxb, 0x00800000u /*enc(-FLT_MAX)*/, nh);
    gat_zero_f32<<<(unsigned)((nh + 255) / 256), 256, 0, stream>>>(sumb, nh);

    gat_logits<<<ET / 8, 256, 0, stream>>>(xl, xr, e_emb, eidx,
                                           att + (size_t)l * HH * CC, num, maxb);
    long eh = (long)ET * HH;
    gat_expsum<<<(unsigned)((eh + 255) / 256), 256, 0, stream>>>(eidx, num, maxb, sumb);

    long nf = (long)NN * FF;
    gat_zero_f32<<<(unsigned)((nf + 255) / 256), 256, 0, stream>>>(agg, nf);
    gat_aggregate<<<ET, FF, 0, stream>>>(eidx, num, sumb, xl, agg);

    gat_ln<<<NN, FF, 0, stream>>>(x, agg, gat_bias + (size_t)l * FF,
                                  ln1_g + (size_t)l * FF, ln1_b + (size_t)l * FF, hbuf);
    launch_gemm(hbuf, B1 + (size_t)l * F2 * FF, nullptr, mid, NN, F2, FF, 2, stream); // GELU
    launch_gemm(mid, B2 + (size_t)l * FF * F2, nullptr, ybuf, NN, FF, F2, 0, stream);
    gat_ln<<<NN, FF, 0, stream>>>(hbuf, ybuf, nullptr,
                                  ln2_g + (size_t)l * FF, ln2_b + (size_t)l * FF, x);
  }

  // ---- final copy x -> d_out through the Tensor Data Mover ----
  gat_tdm_copy<<<(NN * FF) / TDM_CHUNK, 256, 0, stream>>>(x, (float*)d_out);
}
